// QuantizedLinear_19920058318978
// MI455X (gfx1250) — matrix operand order repaired and checked by reading, not compiled, not run on hardware
//
#include <hip/hip_runtime.h>

// ---------------------------------------------------------------------------
// W4A16 quantized GEMM for gfx1250 (MI455X):
//   out[m,n] = f16( sum_k a[m,k] * (f16)w_q[k,n] * scale[k/128, n] )  (f32 acc)
// Compute-bound (137 GFLOP vs ~5.5us HBM traffic at 23.3 TB/s) -> built around
// v_wmma_f32_16x16x32_f16. A-tiles are staged to LDS by the Tensor Data Mover
// (tensor_load_to_lds + s_wait_tensorcnt); W-tiles are dequantized in-register
// (v_cvt_f16_u16 + packed f16 mul) and stored transposed into LDS.
// ---------------------------------------------------------------------------

typedef __attribute__((ext_vector_type(16))) _Float16 v16h;
typedef __attribute__((ext_vector_type(8)))  _Float16 v8h;
typedef __attribute__((ext_vector_type(8)))  float    v8f;
typedef __attribute__((ext_vector_type(4)))  int      v4i;
typedef __attribute__((ext_vector_type(4)))  unsigned u32x4;
typedef __attribute__((ext_vector_type(8)))  int      i32x8;
typedef __attribute__((ext_vector_type(4)))  int      i32x4;

#define MDIM 4096
#define NDIM 4096
#define KDIM 4096
#define GRP  128

#define BM 128
#define BN 128
#define BK 32
// LDS row strides in halves: 48 halves = 96 B (16-B aligned ds_load_b128,
// bank-spreading). The TDM pad feature reproduces this stride for As.
#define LDA 48
#define LDB 48

#if defined(__has_builtin)
#  if __has_builtin(__builtin_amdgcn_tensor_load_to_lds)
#    define HAVE_TDM 1
#  endif
#endif

#ifdef HAVE_TDM
// Issue a TDM load of one A tile: tile_dim0=32 halves along K (contiguous),
// tile_dim1=128 rows along M, from the 4096x4096 f16 tensor, into LDS at
// lds_off with 32 B of padding appended to each 64 B row (row stride 96 B).
__device__ __forceinline__ void tdm_load_A_tile(const _Float16* gsrc, unsigned lds_off)
{
    const unsigned long long ga = (unsigned long long)(uintptr_t)gsrc;

    u32x4 g0;
    g0[0] = 1u;                                   // count=1, user descriptor
    g0[1] = lds_off;                              // LDS byte address
    g0[2] = (unsigned)(ga & 0xFFFFFFFFu);         // global_addr[31:0]
    g0[3] = (unsigned)((ga >> 32) & 0x01FFFFFFu)  // global_addr[56:32]
          | (2u << 30);                           // type = 2 ("image")

    i32x8 g1;
    g1[0] = (int)((1u << 16)      // data_size = 1 -> 2-byte elements
                | (1u << 20)      // pad_enable
                | (3u << 22)      // pad_interval code 3 -> every 16 DWORDs (64 B)
                | (7u << 25));    // pad_amount  code 7 -> 8 DWORDs (32 B)
    g1[1] = (int)((KDIM & 0xFFFFu) << 16);        // tensor_dim0[15:0] @ bits 63:48
    g1[2] = (int)(((KDIM >> 16) & 0xFFFFu)        // tensor_dim0[31:16]
                | ((MDIM & 0xFFFFu) << 16));      // tensor_dim1[15:0] @ bits 95:80
    g1[3] = (int)(((MDIM >> 16) & 0xFFFFu)        // tensor_dim1[31:16]
                | ((unsigned)BK << 16));          // tile_dim0 = 32 @ bits 127:112
    g1[4] = (int)(BM & 0xFFFFu);                  // tile_dim1 = 128; tile_dim2 = 0
    g1[5] = (int)KDIM;                            // tensor_dim0_stride[31:0] = 4096
    g1[6] = 0;                                    // stride[47:32], dim1_stride lo
    g1[7] = 0;

    i32x4 gz = (i32x4)0;                          // groups 2/3 unused (2-D tile)

#if __has_include(<hip/amd_detail/amd_gfx1250_TDM.h>)
    i32x8 gz8 = (i32x8)0;
    __builtin_amdgcn_tensor_load_to_lds(g0, g1, gz, gz, gz8, 0);   // clang-23 form
#else
    __builtin_amdgcn_tensor_load_to_lds(g0, g1, gz, gz, 0);        // ROCm 7.2 form
#endif
}
#endif // HAVE_TDM

__global__ __launch_bounds__(256)
void qgemm_wmma_f16(const _Float16* __restrict__ A,
                    const int*      __restrict__ Wq,
                    const _Float16* __restrict__ S,
                    _Float16*       __restrict__ Out)
{
    __shared__ _Float16 As[BM * LDA];   // A tile, [m][k], k contiguous, stride 48
    __shared__ _Float16 Bs[BN * LDB];   // dequantized W tile, TRANSPOSED: [n][k]

    const int tid  = threadIdx.x;
    const int lane = tid & 31;
    const int wave = tid >> 5;          // 0..7
    const int wm   = wave >> 1;         // 0..3 -> 32-row slab
    const int wn   = wave & 1;          // 0..1 -> 64-col slab

    const int bm = blockIdx.y * BM;
    const int bn = blockIdx.x * BN;

    // W-tile loader mapping: thread t -> k-row t/8, 16 n's at (t%8)*16
    const int w_k = tid >> 3;
    const int w_n = (tid & 7) * 16;

#ifndef HAVE_TDM
    // Fallback A-tile loader mapping: thread t -> row t/2, 16 halves at (t%2)*16
    const int a_row = tid >> 1;
    const int a_seg = (tid & 1) * 16;
#endif

    const unsigned as_base = (unsigned)(uintptr_t)(&As[0]);
    (void)as_base;

    v8f acc[2][4];
#pragma unroll
    for (int i = 0; i < 2; ++i)
#pragma unroll
        for (int j = 0; j < 4; ++j)
            acc[i][j] = (v8f)0.0f;

    for (int kg = 0; kg < KDIM / GRP; ++kg) {
        // Scales are constant over a whole K-group (4 consecutive BK tiles):
        // hoist the 32-byte scale vector load out of the K-step loop.
        const _Float16* sg = S + (size_t)kg * NDIM + (bn + w_n);
        v8h s0 = *(const v8h*)(sg);
        v8h s1 = *(const v8h*)(sg + 8);
        _Float16 sc[16];
        *(v8h*)&sc[0] = s0;
        *(v8h*)&sc[8] = s1;

        for (int kt = 0; kt < GRP / BK; ++kt) {
            const int k0 = kg * GRP + kt * BK;

            // ---------------- stage A tile: 128x32 halves -----------------
#ifdef HAVE_TDM
            if (tid < 32) {     // one wave drives the Tensor Data Mover
                tdm_load_A_tile(A + (size_t)bm * KDIM + k0, as_base);
            }
#else
            {
                const _Float16* ag = A + (size_t)(bm + a_row) * KDIM + (k0 + a_seg);
                v8h a0 = *(const v8h*)(ag);
                v8h a1 = *(const v8h*)(ag + 8);
                *(v8h*)(&As[a_row * LDA + a_seg])     = a0;
                *(v8h*)(&As[a_row * LDA + a_seg + 8]) = a1;
            }
#endif

            // -------- stage + dequantize W tile (transposed into Bs) -------
            const int* wg = Wq + (size_t)(k0 + w_k) * NDIM + (bn + w_n);
            v4i q0 = *(const v4i*)(wg);
            v4i q1 = *(const v4i*)(wg + 4);
            v4i q2 = *(const v4i*)(wg + 8);
            v4i q3 = *(const v4i*)(wg + 12);

            int q[16];
            *(v4i*)&q[0]  = q0;  *(v4i*)&q[4]  = q1;
            *(v4i*)&q[8]  = q2;  *(v4i*)&q[12] = q3;

#pragma unroll
            for (int i = 0; i < 16; ++i) {
                // values are in [0,16): u16 -> f16 is exact (v_cvt_f16_u16)
                _Float16 w = (_Float16)(unsigned short)(unsigned)q[i] * sc[i];
                Bs[(size_t)(w_n + i) * LDB + w_k] = w;   // transposed [n][k]
            }

            // Prefetch next W tile while this one is consumed.
            if (k0 + BK < KDIM) {
                __builtin_prefetch(wg + (size_t)BK * NDIM, 0, 0);
            }

#ifdef HAVE_TDM
            if (tid < 32) {
                __builtin_amdgcn_s_wait_tensorcnt((short)0);
            }
#endif
            __syncthreads();

            // ------------------ load fragments from LDS -------------------
            // A 16x32 f16 fragment (ISA 7.12.2): lane L holds row M=L%16;
            // VGPR0-3: K=base..base+7, VGPR4-7: K=base+16..base+23,
            // base = 0 (lanes 0-15) / 8 (lanes 16-31).
            v16h afrag[2];
#pragma unroll
            for (int mt = 0; mt < 2; ++mt) {
                const int row = wm * 32 + mt * 16 + (lane & 15);
                const int kb  = (lane < 16) ? 0 : 8;
                v8h lo = *(const v8h*)(&As[row * LDA + kb]);
                v8h hi = *(const v8h*)(&As[row * LDA + kb + 16]);
                v16h f;
#pragma unroll
                for (int i = 0; i < 8; ++i) { f[i] = lo[i]; f[i + 8] = hi[i]; }
                afrag[mt] = f;
            }

            // B 32x16 f16 fragment: lane L holds column N=L%16; same K order as A:
            // VGPR0-3: K=base..base+7, VGPR4-7: K=base+16..base+23,
            // base = 0 (lanes 0-15) / 8 (lanes 16-31) -> two aligned 16-byte ds loads.
            v16h bfrag[4];
#pragma unroll
            for (int nt = 0; nt < 4; ++nt) {
                const int col = wn * 64 + nt * 16 + (lane & 15);
                const int kb  = (lane < 16) ? 0 : 8;
                v8h lo = *(const v8h*)(&Bs[col * LDB + kb]);
                v8h hi = *(const v8h*)(&Bs[col * LDB + kb + 16]);
                v16h f;
#pragma unroll
                for (int i = 0; i < 8; ++i) { f[i] = lo[i]; f[i + 8] = hi[i]; }
                bfrag[nt] = f;
            }

            // -------------------------- 8 WMMAs ---------------------------
#pragma unroll
            for (int mt = 0; mt < 2; ++mt)
#pragma unroll
                for (int nt = 0; nt < 4; ++nt)
                    acc[mt][nt] = __builtin_amdgcn_wmma_f32_16x16x32_f16(
                        /*neg_a=*/false, afrag[mt],
                        /*neg_b=*/false, bfrag[nt],
                        /*c_mod=*/(short)0, acc[mt][nt],
                        /*reuse_a=*/false, /*reuse_b=*/false);

            __syncthreads();
        }
    }

    // ------------------------------ epilogue ------------------------------
    // C/D 16x16 f32 layout: VGPR r holds M=r (lanes 0-15) / M=r+8 (lanes 16-31),
    // N = lane % 16.
#pragma unroll
    for (int mt = 0; mt < 2; ++mt) {
#pragma unroll
        for (int nt = 0; nt < 4; ++nt) {
            const int col   = bn + wn * 64 + nt * 16 + (lane & 15);
            const int rbase = bm + wm * 32 + mt * 16 + ((lane < 16) ? 0 : 8);
#pragma unroll
            for (int r = 0; r < 8; ++r) {
                Out[(size_t)(rbase + r) * NDIM + col] = (_Float16)acc[mt][nt][r];
            }
        }
    }
}

extern "C" void kernel_launch(void* const* d_in, const int* in_sizes, int n_in,
                              void* d_out, int out_size, void* d_ws, size_t ws_size,
                              hipStream_t stream) {
    (void)in_sizes; (void)n_in; (void)out_size; (void)d_ws; (void)ws_size;
    const _Float16* a  = (const _Float16*)d_in[0];  // fp16 activations [M,K]
    const int*      wq = (const int*)d_in[1];       // int32-held 4-bit weights [K,N]
    const _Float16* s  = (const _Float16*)d_in[2];  // fp16 scales [K/G, N]
    // d_in[3] = group_size scalar (128), baked into the kernel.
    _Float16* out = (_Float16*)d_out;               // fp16 output [M,N]

    dim3 grid(NDIM / BN, MDIM / BM);  // (32, 32)
    dim3 block(256);
    qgemm_wmma_f16<<<grid, block, 0, stream>>>(a, wq, s, out);
}
